// MultiGeometryAttention_953482739859
// MI455X (gfx1250) — compile-verified, matrix operand order checked by reading, not run on hardware
//
#include <hip/hip_runtime.h>

// ---------------------------------------------------------------------------
// MultiGeometryAttention for MI455X (gfx1250): bf16 WMMA everywhere.
// B=2, T=2048, C=1024, H=16, HD=64, N1=6 (sdpa), N2=4 (hyperbolic), N3=6 (cos)
// ---------------------------------------------------------------------------

typedef __attribute__((ext_vector_type(16))) __bf16 v16bf;
typedef __attribute__((ext_vector_type(8)))  float  v8f;

struct alignas(16) U4 { unsigned int x, y, z, w; };
union AB { v16bf v; U4 u[2]; };

static __device__ inline v8f wmma_bf16(v16bf a, v16bf b, v8f c) {
  // D = A(16x32) * B(32x16) + C(16x16), fp32 accumulate
  return __builtin_amdgcn_wmma_f32_16x16x32_bf16(
      /*neg_a=*/false, a, /*neg_b=*/false, b,
      /*c_mod=*/(short)0, c, /*reuse_a=*/false, /*reuse_b=*/false);
}

#define Bb 2
#define Tt 2048
#define Cc 1024
#define Hh 16
#define HD 64
#define N1h 6
#define N2h 4

// ---------------------------------------------------------------------------
// 1) fp32 -> bf16 conversion
// ---------------------------------------------------------------------------
__global__ void f32_to_bf16_k(const float* __restrict__ in, __bf16* __restrict__ out, int n) {
  int i = blockIdx.x * blockDim.x + threadIdx.x;
  if (i < n) out[i] = (__bf16)in[i];
}

// ---------------------------------------------------------------------------
// 2/5) bf16 GEMM: Cout[M,N] = A[M,K] * Bw[N,K]^T (+ bias).
//      One wave computes a 32x64 tile (8 WMMAs / 4 B-operands per K-step).
//      Register double-buffering: next K-step's operands are in flight while
//      the current 8 WMMAs execute (waits become s_wait_loadcnt N>0).
//      Blocks: 8 waves stacked in M sharing the same 64-wide B panel.
// ---------------------------------------------------------------------------
__global__ void __launch_bounds__(256) gemm_bf16_k(
    const __bf16* __restrict__ A, const __bf16* __restrict__ Bw,
    float* __restrict__ Cout, const float* __restrict__ bias,
    int M, int N, int K) {
  const int lane  = threadIdx.x & 31;
  const int wslot = threadIdx.x >> 5;                 // 0..7 within block
  const int tilesN = N >> 6;
  const int n0 = (blockIdx.x % tilesN) << 6;
  const int m0 = ((blockIdx.x / tilesN) << 8) + (wslot << 5);
  if (m0 >= M) return;

  const int lrow = lane & 15;           // A row / B col within 16-tile
  const int kb   = (lane >> 4) << 3;    // K sub-block: 0 or 8
  const int half = lane >> 4;

  v8f acc[2][4];
#pragma unroll
  for (int g = 0; g < 2; ++g)
#pragma unroll
    for (int t = 0; t < 4; ++t)
#pragma unroll
      for (int r = 0; r < 8; ++r) acc[g][t][r] = 0.0f;

  const __bf16* a0p = A + (size_t)(m0 + lrow) * K + kb;
  const __bf16* a1p = A + (size_t)(m0 + 16 + lrow) * K + kb;
  const __bf16* bp0 = Bw + (size_t)(n0 + lrow) * K + kb;

  auto loadstep = [&](int kO, AB* av, AB* bv) {
    const __bf16* p0 = a0p + kO;
    av[0].u[0] = *(const U4*)(p0);
    av[0].u[1] = *(const U4*)(p0 + 16);
    const __bf16* p1 = a1p + kO;
    av[1].u[0] = *(const U4*)(p1);
    av[1].u[1] = *(const U4*)(p1 + 16);
#pragma unroll
    for (int t = 0; t < 4; ++t) {
      const __bf16* bp = bp0 + (size_t)(16 * t) * K + kO;
      bv[t].u[0] = *(const U4*)(bp);
      bv[t].u[1] = *(const U4*)(bp + 16);
    }
  };

  AB a[2], bb[4];
  loadstep(0, a, bb);

  for (int k0 = 0; k0 < K - 32; k0 += 32) {
    AB an[2], bn[4];
    loadstep(k0 + 32, an, bn);            // in flight during WMMAs below
    if (k0 + 64 < K) {                    // pull k+64 lines toward the WGP
      __builtin_prefetch(a0p + k0 + 64, 0, 1);
      __builtin_prefetch(a1p + k0 + 64, 0, 1);
#pragma unroll
      for (int t = 0; t < 4; ++t)
        __builtin_prefetch(bp0 + (size_t)(16 * t) * K + k0 + 64, 0, 1);
    }
#pragma unroll
    for (int t = 0; t < 4; ++t) {
      acc[0][t] = wmma_bf16(a[0].v, bb[t].v, acc[0][t]);
      acc[1][t] = wmma_bf16(a[1].v, bb[t].v, acc[1][t]);
    }
#pragma unroll
    for (int i = 0; i < 2; ++i) a[i] = an[i];
#pragma unroll
    for (int t = 0; t < 4; ++t) bb[t] = bn[t];
  }
#pragma unroll
  for (int t = 0; t < 4; ++t) {
    acc[0][t] = wmma_bf16(a[0].v, bb[t].v, acc[0][t]);
    acc[1][t] = wmma_bf16(a[1].v, bb[t].v, acc[1][t]);
  }

  // C/D layout: VGPR r, lanes 0-15 -> M=r, lanes 16-31 -> M=r+8; N = lane&15
#pragma unroll
  for (int g = 0; g < 2; ++g) {
    const int mbase = m0 + 16 * g + 8 * half;
#pragma unroll
    for (int t = 0; t < 4; ++t) {
      const int n = n0 + 16 * t + lrow;
      const float bv = bias ? bias[n] : 0.0f;
#pragma unroll
      for (int r = 0; r < 8; ++r)
        Cout[(size_t)(mbase + r) * N + n] = acc[g][t][r] + bv;
    }
  }
}

// ---------------------------------------------------------------------------
// 3) Rotary + split + per-geometry prep. One wave per (b,h,t) row.
//    Lane i owns the rotary pair (d=i, d=i+32).
//    Outputs: qh/kh [b,h,t,d] bf16, vt [b,h,d,t] bf16 (transposed),
//             tq/tk [b, h-N1, t] fp32 for hyperbolic heads.
// ---------------------------------------------------------------------------
__global__ void __launch_bounds__(256) rotary_split_k(
    const float* __restrict__ qkv, const float* __restrict__ curv,
    __bf16* __restrict__ qh, __bf16* __restrict__ kh, __bf16* __restrict__ vt,
    float* __restrict__ tq, float* __restrict__ tk) {
  const int lane = threadIdx.x & 31;
  const int wid  = (int)((blockIdx.x * blockDim.x + threadIdx.x) >> 5);
  const int t = wid & (Tt - 1);
  const int h = (wid >> 11) & (Hh - 1);
  const int b = wid >> 15;

  const float* base = qkv + (size_t)(b * Tt + t) * (3 * Cc) + h * HD;
  float q1 = base[lane],          q2 = base[lane + 32];
  float k1 = base[Cc + lane],     k2 = base[Cc + lane + 32];
  float v1 = base[2 * Cc + lane], v2 = base[2 * Cc + lane + 32];

  // inv_freq[i] = 10000^(-i/32), angle = t * inv_freq
  float freq = __powf(10000.0f, -(float)lane * (1.0f / 32.0f));
  float ang = (float)t * freq;
  float sn, cs;
  __sincosf(ang, &sn, &cs);
  float qr1 = q1 * cs + q2 * sn, qr2 = -q1 * sn + q2 * cs;
  float kr1 = k1 * cs + k2 * sn, kr2 = -k1 * sn + k2 * cs;

  float sq = qr1 * qr1 + qr2 * qr2;
  float sk = kr1 * kr1 + kr2 * kr2;
#pragma unroll
  for (int m = 1; m < 32; m <<= 1) {
    sq += __shfl_xor(sq, m, 32);
    sk += __shfl_xor(sk, m, 32);
  }

  if (h >= N1h && h < N1h + N2h) {
    float c = curv[h - N1h];
    float tqv = sqrtf(1.0f / c + sq);
    float tkv = sqrtf(1.0f / c + sk);
    if (lane == 0) {
      size_t o = (size_t)(b * N2h + (h - N1h)) * Tt + t;
      tq[o] = tqv;
      tk[o] = tkv;
    }
  } else if (h >= N1h + N2h) {
    float iq = 1.0f / fmaxf(sqrtf(sq), 1e-12f);
    float ik = 1.0f / fmaxf(sqrtf(sk), 1e-12f);
    qr1 *= iq; qr2 *= iq;
    kr1 *= ik; kr2 *= ik;
  }

  const size_t ro = ((size_t)(b * Hh + h) * Tt + t) * HD;
  qh[ro + lane] = (__bf16)qr1; qh[ro + lane + 32] = (__bf16)qr2;
  kh[ro + lane] = (__bf16)kr1; kh[ro + lane + 32] = (__bf16)kr2;
  const size_t vo = (size_t)(b * Hh + h) * HD;
  vt[(vo + lane) * Tt + t]      = (__bf16)v1;
  vt[(vo + lane + 32) * Tt + t] = (__bf16)v2;
}

// ---------------------------------------------------------------------------
// 4) Flash attention. One wave per (b,h, 16-row Q tile); 32-key KV chunks.
//    mode 0: scores*1/8   mode 1: hyperbolic   mode 2: cosine (scores*8)
// ---------------------------------------------------------------------------
__global__ void __launch_bounds__(128) attn_flash_k(
    const __bf16* __restrict__ qh, const __bf16* __restrict__ kh,
    const __bf16* __restrict__ vt, const float* __restrict__ tq,
    const float* __restrict__ tk, const float* __restrict__ curv,
    __bf16* __restrict__ yb) {
  const int lane = threadIdx.x & 31;
  const int wv   = (threadIdx.x >> 5) & 3;
  const int wid  = (int)((blockIdx.x * blockDim.x + threadIdx.x) >> 5);
  const int bh = wid >> 7;             // (b*H + h)
  const int t0 = (wid & 127) << 4;     // Q tile start
  const int h = bh & (Hh - 1), b = bh >> 4;

  __shared__ alignas(16) __bf16 pbuf[4][16 * 32];

  const int lrow = lane & 15;
  const int kb   = (lane >> 4) << 3;   // 0 or 8
  const int half = lane >> 4;

  // Q A-operands for the two 32-wide d chunks (stay in registers)
  const __bf16* qbase = qh + ((size_t)bh * Tt + t0) * HD;
  AB qa[2];
#pragma unroll
  for (int c = 0; c < 2; ++c) {
    const __bf16* p = qbase + (size_t)lrow * HD + 32 * c + kb;
    qa[c].u[0] = *(const U4*)(p);
    qa[c].u[1] = *(const U4*)(p + 16);
  }

  const int mode = (h < N1h) ? 0 : (h < N1h + N2h ? 1 : 2);
  const float scale = (h < N1h) ? 0.125f : 8.0f;
  float cv = 1.0f, srct = 1.0f;
  float tqr[8];
  const float* tkb = nullptr;
  if (mode == 1) {
    cv = curv[h - N1h];
    srct = sqrtf(1.0f / cv);
    const float* tqb = tq + (size_t)(b * N2h + (h - N1h)) * Tt;
    tkb = tk + (size_t)(b * N2h + (h - N1h)) * Tt;
#pragma unroll
    for (int r = 0; r < 8; ++r) tqr[r] = tqb[t0 + r + 8 * half];
  }

  v8f o[4];
  float mi[8], li[8];
#pragma unroll
  for (int t = 0; t < 4; ++t)
#pragma unroll
    for (int r = 0; r < 8; ++r) o[t][r] = 0.0f;
#pragma unroll
  for (int r = 0; r < 8; ++r) { mi[r] = -1e30f; li[r] = 0.0f; }

  const __bf16* kbase = kh + (size_t)bh * Tt * HD;
  const __bf16* vbase = vt + (size_t)bh * HD * Tt;

  for (int kk = 0; kk < t0 + 16; kk += 32) {
    // prefetch next KV chunk while this one is consumed
    if (kk + 32 < t0 + 16) {
      __builtin_prefetch(kbase + (size_t)(kk + 32 + lrow) * HD, 0, 1);
      __builtin_prefetch(kbase + (size_t)(kk + 48 + lrow) * HD, 0, 1);
      __builtin_prefetch(vbase + (size_t)lrow * Tt + kk + 32, 0, 1);
      __builtin_prefetch(vbase + (size_t)(32 + lrow) * Tt + kk + 32, 0, 1);
    }

    // ---- scores: S(16x32) via 4 WMMAs --------------------------------
    v8f s[2];
#pragma unroll
    for (int j = 0; j < 2; ++j) {
      const int key = kk + 16 * j + lrow;
      const __bf16* kp = kbase + (size_t)key * HD + kb;
      AB b0, b1;
      b0.u[0] = *(const U4*)(kp);       b0.u[1] = *(const U4*)(kp + 16);
      b1.u[0] = *(const U4*)(kp + 32);  b1.u[1] = *(const U4*)(kp + 48);
      v8f z;
#pragma unroll
      for (int r = 0; r < 8; ++r) z[r] = 0.0f;
      z = wmma_bf16(qa[0].v, b0.v, z);
      s[j] = wmma_bf16(qa[1].v, b1.v, z);
    }

    // ---- per-mode transform + causal mask + chunk row-max -------------
    float tk0 = 0.0f, tk1 = 0.0f;
    if (mode == 1) { tk0 = tkb[kk + lrow]; tk1 = tkb[kk + 16 + lrow]; }
    float pr0[8], pr1[8], cmax[8];
#pragma unroll
    for (int r = 0; r < 8; ++r) {
      const int trow = t0 + r + 8 * half;
      float s0 = s[0][r], s1 = s[1][r];
      if (mode == 1) {
        float a0 = fmaxf(cv * (tqr[r] * tk0 - s0), 1.0f + 1e-7f);
        float a1 = fmaxf(cv * (tqr[r] * tk1 - s1), 1.0f + 1e-7f);
        s0 = 1.0f / (1e-6f + srct * acoshf(a0));
        s1 = 1.0f / (1e-6f + srct * acoshf(a1));
      } else {
        s0 *= scale; s1 *= scale;
      }
      if (kk + lrow > trow)      s0 = -1e30f;
      if (kk + 16 + lrow > trow) s1 = -1e30f;
      pr0[r] = s0; pr1[r] = s1;
      float mx = fmaxf(s0, s1);
#pragma unroll
      for (int msk = 1; msk < 16; msk <<= 1)
        mx = fmaxf(mx, __shfl_xor(mx, msk, 32));   // stays within lane half
      cmax[r] = mx;
    }

    // ---- online softmax update; stage P into LDS (C-layout -> A-layout)
#pragma unroll
    for (int r = 0; r < 8; ++r) {
      const float newm = fmaxf(mi[r], cmax[r]);
      const float alpha = __expf(mi[r] - newm);
      const float p0 = __expf(pr0[r] - newm);
      const float p1 = __expf(pr1[r] - newm);
      float rs = p0 + p1;
#pragma unroll
      for (int msk = 1; msk < 16; msk <<= 1)
        rs += __shfl_xor(rs, msk, 32);
      li[r] = li[r] * alpha + rs;
      mi[r] = newm;
#pragma unroll
      for (int t = 0; t < 4; ++t) o[t][r] *= alpha;
      const int m = r + 8 * half;
      pbuf[wv][m * 32 + lrow]      = (__bf16)p0;
      pbuf[wv][m * 32 + 16 + lrow] = (__bf16)p1;
    }

    // ---- read P back in A-operand layout (wave-private, in-order DS) --
    AB pa;
    {
      const __bf16* pp = &pbuf[wv][lrow * 32 + kb];
      pa.u[0] = *(const U4*)(pp);
      pa.u[1] = *(const U4*)(pp + 16);
    }

    // ---- O += P(16x32) * V(32x64): 4 WMMAs, V^T gives contiguous loads
#pragma unroll
    for (int t = 0; t < 4; ++t) {
      const int d = 16 * t + lrow;
      const __bf16* vp = vbase + (size_t)d * Tt + kk + kb;
      AB vb;
      vb.u[0] = *(const U4*)(vp);
      vb.u[1] = *(const U4*)(vp + 16);
      o[t] = wmma_bf16(pa.v, vb.v, o[t]);
    }
  }

  // ---- epilogue: normalize by l, write y[b,t, h*64+d] as bf16 ----------
#pragma unroll
  for (int r = 0; r < 8; ++r) {
    const float inv = 1.0f / li[r];
    const int trow = t0 + r + 8 * half;
    __bf16* yp = yb + (size_t)(b * Tt + trow) * Cc + h * HD;
#pragma unroll
    for (int t = 0; t < 4; ++t)
      yp[16 * t + lrow] = (__bf16)(o[t][r] * inv);
  }
}

// ---------------------------------------------------------------------------
// Host launcher
// ---------------------------------------------------------------------------
extern "C" void kernel_launch(void* const* d_in, const int* in_sizes, int n_in,
                              void* d_out, int out_size, void* d_ws, size_t ws_size,
                              hipStream_t stream) {
  (void)in_sizes; (void)n_in; (void)out_size; (void)ws_size;
  const float* x      = (const float*)d_in[0];  // [B,T,C]
  const float* qkv_w  = (const float*)d_in[1];  // [3C,C]
  const float* proj_w = (const float*)d_in[2];  // [C,C]
  const float* proj_b = (const float*)d_in[3];  // [C]
  const float* curv   = (const float*)d_in[4];  // [N2]
  float* out = (float*)d_out;                   // [B,T,C]

  const size_t nX = (size_t)Bb * Tt * Cc;       // 4,194,304
  const size_t nWq = (size_t)3 * Cc * Cc;       // 3,145,728
  const size_t nWp = (size_t)Cc * Cc;           // 1,048,576

  char* w = (char*)d_ws;
  __bf16* xb     = (__bf16*)w; w += nX * 2;
  __bf16* wqkvb  = (__bf16*)w; w += nWq * 2;
  __bf16* wprojb = (__bf16*)w; w += nWp * 2;
  float*  qkvf   = (float*)w;  w += (size_t)Bb * Tt * 3 * Cc * 4;
  __bf16* qhb    = (__bf16*)w; w += nX * 2;
  __bf16* khb    = (__bf16*)w; w += nX * 2;
  __bf16* vtb    = (__bf16*)w; w += nX * 2;
  __bf16* ybb    = (__bf16*)w; w += nX * 2;
  float*  tqa    = (float*)w;  w += (size_t)Bb * N2h * Tt * 4;
  float*  tka    = (float*)w;  w += (size_t)Bb * N2h * Tt * 4;

  f32_to_bf16_k<<<(int)(nX  / 256), 256, 0, stream>>>(x, xb, (int)nX);
  f32_to_bf16_k<<<(int)(nWq / 256), 256, 0, stream>>>(qkv_w, wqkvb, (int)nWq);
  f32_to_bf16_k<<<(int)(nWp / 256), 256, 0, stream>>>(proj_w, wprojb, (int)nWp);

  // qkv = x @ qkv_w^T : M=4096, N=3072, K=1024
  // blocks = (M/256) * (N/64) = 16 * 48 = 768, 8 waves each (32x64 per wave)
  gemm_bf16_k<<<768, 256, 0, stream>>>(xb, wqkvb, qkvf, nullptr, Bb * Tt, 3 * Cc, Cc);

  // rotary + split: B*H*T = 65536 waves / 8 per block
  rotary_split_k<<<8192, 256, 0, stream>>>(qkvf, curv, qhb, khb, vtb, tqa, tka);

  // attention: B*H*(T/16) = 4096 waves / 4 per block
  attn_flash_k<<<1024, 128, 0, stream>>>(qhb, khb, vtb, tqa, tka, curv, ybb);

  // out = y @ proj_w^T + b : M=4096, N=1024, K=1024 -> 16 * 16 = 256 blocks
  gemm_bf16_k<<<256, 256, 0, stream>>>(ybb, wprojb, out, proj_b, Bb * Tt, Cc, Cc);
}
